// Document_embedder_58884001628714
// MI455X (gfx1250) — compile-verified, matrix operand order checked by reading, not run on hardware
//
#include <hip/hip_runtime.h>
#include <math.h>

// ---------------------------------------------------------------------------
// Problem constants (B,T,D,U) = (64,1024,512,512), Keras GRU reset_after=True.
// ---------------------------------------------------------------------------
constexpr int Bc = 64, Tc = 1024, Dc = 512, Uc = 512;
constexpr int BT = Bc * Tc;      // 65536 GEMM rows
constexpr int N3 = 3 * Uc;       // 1536 gate columns

typedef __attribute__((ext_vector_type(16))) __bf16 v16bf;
typedef __attribute__((ext_vector_type(8)))  float  v8f;
typedef __attribute__((ext_vector_type(4)))  __bf16 v4bf;
typedef __attribute__((ext_vector_type(4)))  unsigned int u32x4;
typedef __attribute__((ext_vector_type(8)))  int i32x8;
typedef __attribute__((ext_vector_type(4)))  int i32x4;

union Frag { v16bf v; uint4 q[2]; };

__device__ inline v8f wmma_bf16(const Frag& a, const Frag& b, v8f c) {
    // D = A(16x32 bf16) x B(32x16 bf16) + C(16x16 f32)
    return __builtin_amdgcn_wmma_f32_16x16x32_bf16(
        /*neg_a=*/false, a.v, /*neg_b=*/false, b.v,
        /*c_mod=*/(short)0, c, /*reuse_a=*/false, /*reuse_b=*/false);
}

// ---------------------------------------------------------------------------
// TDM: async-DMA a 2D bf16 tile (tile_rows x tile_cols) from a row-major
// matrix (row stride = stride_elems) into LDS at byte offset lds_off.
// LDS padding: +4 DWORDs (8 bf16) after every 32 DWORDs (64 bf16) -> the
// destination row pitch is 72 bf16, matching the bank-conflict-free layout
// the WMMA fragment reads use.  D# packing per CDNA5 ISA.
// ---------------------------------------------------------------------------
__device__ inline void tdm_load_tile_2d(unsigned lds_off, const void* gsrc,
                                        unsigned tile_cols, unsigned tile_rows,
                                        unsigned stride_elems) {
    unsigned long long ga = (unsigned long long)(uintptr_t)gsrc;
    u32x4 g0 = { 1u,                                  // count=1, user desc
                 lds_off,                             // lds_addr (bytes)
                 (unsigned)(ga & 0xFFFFFFFFu),
                 (unsigned)((ga >> 32) & 0x01FFFFFFull) | 0x80000000u }; // type=2
    const unsigned tdim0 = stride_elems;              // tensor dim0 (elements)
    const unsigned tdim1 = 1u << 16;                  // generous row bound
    i32x8 g1;
    g1[0] = (int)((1u << 16)      // data_size = 2 bytes
                | (1u << 20)      // pad_enable
                | (4u << 22)      // pad_interval code 4 = 32 DWORDs (64 bf16)
                | (3u << 25));    // pad_amount  code 3 =  4 DWORDs ( 8 bf16)
    g1[1] = (int)((tdim0 & 0xFFFFu) << 16);                              // dim0 lo
    g1[2] = (int)(((tdim0 >> 16) & 0xFFFFu) | ((tdim1 & 0xFFFFu) << 16));// dim0 hi|dim1 lo
    g1[3] = (int)(((tdim1 >> 16) & 0xFFFFu) | ((tile_cols & 0xFFFFu) << 16)); // dim1 hi|tile0
    g1[4] = (int)(tile_rows & 0xFFFFu);               // tile_dim1 | tile_dim2=0
    g1[5] = (int)tdim0;                               // tensor_dim0_stride lo
    g1[6] = 0;                                        // stride0 hi | stride1 lo
    g1[7] = 0;
    i32x4 z4 = {0, 0, 0, 0};                          // 2D tile: groups 2/3 unused
#if __clang_major__ >= 23
    i32x8 z8 = {0, 0, 0, 0, 0, 0, 0, 0};
    __builtin_amdgcn_tensor_load_to_lds(g0, g1, z4, z4, z8, 0);
#else
    __builtin_amdgcn_tensor_load_to_lds(g0, g1, z4, z4, 0);
#endif
}

// ---------------------------------------------------------------------------
// Workspace layout (bytes, all regions 256B aligned by construction)
// ---------------------------------------------------------------------------
constexpr size_t OFF_XW  = 0;                               // BT x N3 fp32
constexpr size_t SZ_XW   = (size_t)BT * N3 * 4;             // 402.6 MB
constexpr size_t OFF_XB  = OFF_XW + SZ_XW;                  // BT x D bf16
constexpr size_t SZ_XB   = (size_t)BT * Dc * 2;             // 67.1 MB
constexpr size_t OFF_WT  = OFF_XB + SZ_XB;                  // [N3][D] bf16 (W transposed)
constexpr size_t SZ_WT   = (size_t)N3 * Dc * 2;             // 1.5 MB
constexpr size_t OFF_RT  = OFF_WT + SZ_WT;                  // [N3][U] bf16 (R transposed)
constexpr size_t SZ_RT   = (size_t)N3 * Uc * 2;             // 1.5 MB
constexpr size_t OFF_H32 = OFF_RT + SZ_RT;                  // [B][U] fp32 running state
constexpr size_t SZ_H32  = (size_t)Bc * Uc * 4;
constexpr size_t OFF_HBF = OFF_H32 + SZ_H32;                // 2 x [B][U] bf16 (double buffer)
constexpr size_t SZ_HBF  = (size_t)2 * Bc * Uc * 2;
constexpr size_t OFF_CTR = OFF_HBF + SZ_HBF;                // barrier counter

// ---------------------------------------------------------------------------
// Prep kernels
// ---------------------------------------------------------------------------
__global__ void cvt_f32_to_bf16(const float* __restrict__ s, __bf16* __restrict__ d, long n4) {
    long i = (long)blockIdx.x * blockDim.x + threadIdx.x;
    long stride = (long)gridDim.x * blockDim.x;
    const float4* s4 = (const float4*)s;
    v4bf* d4 = (v4bf*)d;
    for (; i < n4; i += stride) {
        float4 f = s4[i];
        v4bf o;
        o[0] = (__bf16)f.x; o[1] = (__bf16)f.y; o[2] = (__bf16)f.z; o[3] = (__bf16)f.w;
        d4[i] = o;
    }
}

// s: [K][N] fp32 (row-major), d: [N][K] bf16  (weights become B-operand [N][K])
__global__ void transpose_to_bf16(const float* __restrict__ s, __bf16* __restrict__ d,
                                  int K, int N) {
    long total = (long)K * N;
    long i = (long)blockIdx.x * blockDim.x + threadIdx.x;
    long stride = (long)gridDim.x * blockDim.x;
    for (; i < total; i += stride) {
        long n = i / K, k = i - n * K;
        d[i] = (__bf16)s[k * (long)N + n];    // coalesced writes on d
    }
}

__global__ void scan_init(float* __restrict__ h32, __bf16* __restrict__ hbf,
                          unsigned* __restrict__ ctr) {
    int i = blockIdx.x * blockDim.x + threadIdx.x;
    int stride = gridDim.x * blockDim.x;
    const int n = Bc * Uc;
    for (int k = i; k < n; k += stride)      h32[k] = 0.0f;
    for (int k = i; k < 2 * n; k += stride)  hbf[k] = (__bf16)0.0f;
    if (i == 0) *ctr = 0u;
}

// ---------------------------------------------------------------------------
// Phase 1: xw = x @ W + b_in    (M=65536, N=1536, K=512) bf16 WMMA, fp32 acc.
// Block tile 64x64.  K-chunk 64, TDM double-buffered: wave 0 issues
// tensor_load_to_lds for chunk kc+1 while all 8 waves compute chunk kc.
// LDS pitch 72 bf16 (TDM pad) -> conflict-free fragment reads.
// 8 waves: wave w -> M-tile (w&3), N-tile pair ((w>>2)*2, +1).
// ---------------------------------------------------------------------------
constexpr int PITCH = 72;                // 64 + 8 bf16 (TDM pad)

__global__ void __launch_bounds__(256)
gemm_xw(const __bf16* __restrict__ X,   // [BT][512] bf16
        const __bf16* __restrict__ Wt,  // [1536][512] bf16
        const float*  __restrict__ bin, // b_in [1536]
        float* __restrict__ xw)         // [BT][1536] fp32
{
    __shared__ __bf16 lA[2][64 * PITCH];
    __shared__ __bf16 lB[2][64 * PITCH];

    const int tid = threadIdx.x;
    const int m0 = blockIdx.x * 64, n0 = blockIdx.y * 64;
    const int l = tid & 31, w = tid >> 5;
    const int lc = l & 15, hi = l >> 4;
    const int m = w & 3, nb = (w >> 2) * 2;

    const __bf16* gA = X + (size_t)m0 * Dc;
    const __bf16* gB = Wt + (size_t)n0 * Dc;
    const unsigned ldsA[2] = { (unsigned)(uintptr_t)&lA[0][0], (unsigned)(uintptr_t)&lA[1][0] };
    const unsigned ldsB[2] = { (unsigned)(uintptr_t)&lB[0][0], (unsigned)(uintptr_t)&lB[1][0] };

    v8f acc0 = {}, acc1 = {};

    if (w == 0) {                                    // preload chunk 0
        tdm_load_tile_2d(ldsA[0], gA, 64, 64, Dc);
        tdm_load_tile_2d(ldsB[0], gB, 64, 64, Dc);
    }

    for (int kc = 0; kc < 8; ++kc) {                 // 8 K-chunks of 64
        if (w == 0) __builtin_amdgcn_s_wait_tensorcnt(0);  // chunk kc landed
        __syncthreads();                             // LDS visible; prev buffer free
        if (w == 0 && kc < 7) {                      // DMA next chunk, overlapped
            const int p = (kc + 1) & 1;
            tdm_load_tile_2d(ldsA[p], gA + (kc + 1) * 64, 64, 64, Dc);
            tdm_load_tile_2d(ldsB[p], gB + (kc + 1) * 64, 64, 64, Dc);
        }
        const __bf16* As = &lA[kc & 1][0];
        const __bf16* Bs = &lB[kc & 1][0];
        for (int kf = 0; kf < 2; ++kf) {
            const int kk = kf * 32 + 8 * hi;         // ISA 16-bit A/B fragment K pattern
            Frag a, b0, b1;
            const __bf16* ap = As + (m * 16 + lc) * PITCH + kk;
            a.q[0] = *(const uint4*)ap;        a.q[1] = *(const uint4*)(ap + 16);
            const __bf16* bp0 = Bs + (nb * 16 + lc) * PITCH + kk;
            b0.q[0] = *(const uint4*)bp0;      b0.q[1] = *(const uint4*)(bp0 + 16);
            const __bf16* bp1 = bp0 + 16 * PITCH;
            b1.q[0] = *(const uint4*)bp1;      b1.q[1] = *(const uint4*)(bp1 + 16);
            acc0 = wmma_bf16(a, b0, acc0);
            acc1 = wmma_bf16(a, b1, acc1);
        }
    }

    const int col0 = n0 + nb * 16 + lc, col1 = col0 + 16;
    const float bi0 = bin[col0], bi1 = bin[col1];
    for (int i = 0; i < 8; ++i) {
        int gr = m0 + m * 16 + i + 8 * hi;           // f32 C-layout: row = i + 8*hi
        xw[(size_t)gr * N3 + col0] = acc0[i] + bi0;
        xw[(size_t)gr * N3 + col1] = acc1[i] + bi1;
    }
}

// ---------------------------------------------------------------------------
// Phase 2: persistent recurrent scan. 16 WGs x 256 threads.
// WG c owns units [32c, 32c+32). Wave w: M-tile m=w&3 (16 batch rows),
// unit-tile uu=w>>2. Each wave: 3 accumulators (z,r,hh gates), K=512 (16 frags),
// A = h_bf16 (double-buffered, device-wide), B = Rt rows (L2-resident).
// One device barrier per step (monotonic counter, release/acquire agent scope).
// ---------------------------------------------------------------------------
__global__ void __launch_bounds__(256)
gru_scan(const float* __restrict__ xw,    // [BT][1536] fp32 (includes b_in)
         const __bf16* __restrict__ Rt,   // [1536][512] bf16
         const float* __restrict__ brec,  // b_rec [1536]
         float* __restrict__ out,         // [B][T][U] fp32
         float* __restrict__ h32,         // [B][U] fp32 running state
         __bf16* __restrict__ hbf,        // 2 x [B][U] bf16
         unsigned* __restrict__ ctr)
{
    const int tid = threadIdx.x, c = blockIdx.x;
    const int l = tid & 31, w = tid >> 5;
    const int lc = l & 15, hi = l >> 4;
    const int m = w & 3, uu = w >> 2;

    const int u    = c * 32 + uu * 16 + lc;   // this lane's unit column
    const int arow = 16 * m + lc;             // A-fragment batch row
    const float bz = brec[u], br = brec[Uc + u], bh = brec[2 * Uc + u];

    // B row pointers are time-invariant: gate g row = g*512 + u
    const __bf16* bR0 = Rt + (size_t)(0 * Uc + u) * Uc;
    const __bf16* bR1 = Rt + (size_t)(1 * Uc + u) * Uc;
    const __bf16* bR2 = Rt + (size_t)(2 * Uc + u) * Uc;

    for (int t = 0; t < Tc; ++t) {
        const __bf16* hr = hbf + (size_t)(t & 1) * (Bc * Uc);
        __bf16*       hw = hbf + (size_t)((t + 1) & 1) * (Bc * Uc);

        // Prefetch this step's xw tile (gfx1250 global_prefetch_b8)
        {
            const float* p = xw + ((size_t)arow * Tc + t) * N3 + (c * 32 + uu * 16);
            __builtin_prefetch(p, 0, 0);
            __builtin_prefetch(p + 512, 0, 0);
            __builtin_prefetch(p + 1024, 0, 0);
        }

        v8f az = {}, ar = {}, ah = {};
        for (int kf = 0; kf < 16; ++kf) {
            const int ko = kf * 32 + 8 * hi;
            Frag a, b0, b1, b2;
            const __bf16* ap = hr + (size_t)arow * Uc + ko;
            a.q[0]  = *(const uint4*)ap;          a.q[1]  = *(const uint4*)(ap + 16);
            b0.q[0] = *(const uint4*)(bR0 + ko);  b0.q[1] = *(const uint4*)(bR0 + ko + 16);
            b1.q[0] = *(const uint4*)(bR1 + ko);  b1.q[1] = *(const uint4*)(bR1 + ko + 16);
            b2.q[0] = *(const uint4*)(bR2 + ko);  b2.q[1] = *(const uint4*)(bR2 + ko + 16);
            az = wmma_bf16(a, b0, az);            // A reused across 3 gates
            ar = wmma_bf16(a, b1, ar);
            ah = wmma_bf16(a, b2, ah);
        }

        // Gate math on this lane's 8 (batch-row, unit) elements
        for (int i = 0; i < 8; ++i) {
            const int b = 16 * m + i + 8 * hi;
            const size_t bt = (size_t)b * Tc + t;
            const float* xp = xw + bt * N3;
            const float xz = xp[u], xr = xp[Uc + u], xh = xp[2 * Uc + u];
            const float z  = 1.0f / (1.0f + __expf(-(xz + az[i] + bz)));
            const float r  = 1.0f / (1.0f + __expf(-(xr + ar[i] + br)));
            const float hh = tanhf(xh + r * (ah[i] + bh));
            const float hp = h32[b * Uc + u];
            const float hn = z * hp + (1.0f - z) * hh;
            h32[b * Uc + u]    = hn;
            out[bt * Uc + u]   = hn;
            hw[b * Uc + u]     = (__bf16)hn;
        }

        // Device-wide barrier (16 WGs): publish h_new, wait for all.
        __threadfence();
        __syncthreads();
        if (tid == 0) {
            __hip_atomic_fetch_add(ctr, 1u, __ATOMIC_RELEASE, __HIP_MEMORY_SCOPE_AGENT);
            const unsigned tgt = 16u * (unsigned)(t + 1);
            while (__hip_atomic_load(ctr, __ATOMIC_ACQUIRE, __HIP_MEMORY_SCOPE_AGENT) < tgt)
                __builtin_amdgcn_s_sleep(2);
        }
        __syncthreads();
        __threadfence();
    }
}

// ---------------------------------------------------------------------------
// Host launcher (graph-capture safe: kernel launches on `stream` only)
// ---------------------------------------------------------------------------
extern "C" void kernel_launch(void* const* d_in, const int* in_sizes, int n_in,
                              void* d_out, int out_size, void* d_ws, size_t ws_size,
                              hipStream_t stream) {
    const float* x    = (const float*)d_in[0];  // [B,T,D]
    const float* W    = (const float*)d_in[1];  // [D,3U]
    const float* R    = (const float*)d_in[2];  // [U,3U]
    const float* bias = (const float*)d_in[3];  // [2,3U]
    float* out = (float*)d_out;

    char* ws = (char*)d_ws;
    float*    xw  = (float*)(ws + OFF_XW);
    __bf16*   xb  = (__bf16*)(ws + OFF_XB);
    __bf16*   Wt  = (__bf16*)(ws + OFF_WT);
    __bf16*   Rt  = (__bf16*)(ws + OFF_RT);
    float*    h32 = (float*)(ws + OFF_H32);
    __bf16*   hbf = (__bf16*)(ws + OFF_HBF);
    unsigned* ctr = (unsigned*)(ws + OFF_CTR);

    // 1) x -> bf16
    cvt_f32_to_bf16<<<4096, 256, 0, stream>>>(x, xb, (long)BT * Dc / 4);
    // 2) W, R -> transposed bf16 [N][K]
    transpose_to_bf16<<<1024, 256, 0, stream>>>(W, Wt, Dc, N3);
    transpose_to_bf16<<<1024, 256, 0, stream>>>(R, Rt, Uc, N3);
    // 3) zero h state + barrier counter (every launch: deterministic replay)
    scan_init<<<64, 256, 0, stream>>>(h32, hbf, ctr);
    // 4) xw = x @ W + b_in  (TDM double-buffered WMMA GEMM)
    gemm_xw<<<dim3(BT / 64, N3 / 64), 256, 0, stream>>>(xb, Wt, bias, xw);
    // 5) recurrent scan (16 persistent WGs, b_rec = bias + 1536)
    gru_scan<<<16, 256, 0, stream>>>(xw, Rt, bias + N3, out, h32, hbf, ctr);
}
